// LSTM_86921548136985
// MI455X (gfx1250) — matrix operand order repaired and checked by reading, not compiled, not run on hardware
//
#include <hip/hip_runtime.h>
#include <math.h>

// ---------------------------------------------------------------------------
// Problem constants (from reference)
// ---------------------------------------------------------------------------
#define WEMB  300
#define CEMB  64
#define HID   512
#define CHID  128
#define TGT   20
#define SEQ   64     // words
#define CPW   16     // chars per word
#define NHASH 4
#define TCH   (SEQ*CPW)       // 1024 char steps
#define GC4   (4*CHID)        // 512  char gate rows
#define GW4   (4*HID)         // 2048 word gate rows
#define XK    448             // padded concat dim: 300 we + 20 zero + 128 hc
#define XPAD  320             // hc starts at col 320

// ---------------------------------------------------------------------------
// WMMA vector types (gfx1250, wave32)
// ---------------------------------------------------------------------------
typedef __attribute__((ext_vector_type(16))) _Float16 v16h;
typedef __attribute__((ext_vector_type(8)))  _Float16 v8h;
typedef __attribute__((ext_vector_type(8)))  float    v8f;

__device__ __forceinline__ float sigm(float x) { return 1.0f / (1.0f + __expf(-x)); }

// ---------------------------------------------------------------------------
// Workspace layout (bytes, all 256B aligned)
// ---------------------------------------------------------------------------
#define OFF_CEBF   0u          // [1024][64]  f16   (gathered char embeds)
#define OFF_WIHC   131072u     // [512][64]   f16   (Wih_c)
#define OFF_XBF    196608u     // [64][448]   f16   (we | 0 | hc)
#define OFF_WIHW   253952u     // [2048][448] f16   (Wih padded)
#define OFF_GC     2088960u    // [1024][512] f32   (char input projections)
#define OFF_GW     4186112u    // [64][2048]  f32   (word input projections)
#define OFF_HFIN   4710400u    // [512]       f32   (final word hidden)

// ---------------------------------------------------------------------------
// Kernel 1: gather + f32->f16 conversion / padding (grid-stride, parallel)
// ---------------------------------------------------------------------------
__global__ void prep_kernel(const int* __restrict__ word_ids,
                            const int* __restrict__ char_ids,
                            const float* __restrict__ word_emb,
                            const float* __restrict__ char_emb,
                            const float* __restrict__ Wih_c,
                            const float* __restrict__ Wih,
                            _Float16* __restrict__ CEbf,
                            _Float16* __restrict__ WihCbf,
                            _Float16* __restrict__ Xbf,
                            _Float16* __restrict__ WihWbf) {
  const int R0 = TCH * CEMB;        // 65536
  const int R1 = GC4 * CEMB;        // 32768
  const int R2 = SEQ * XPAD;        // 20480
  const int R3 = GW4 * XK;          // 917504
  const int NTOT = R0 + R1 + R2 + R3;
  for (int idx = blockIdx.x * blockDim.x + threadIdx.x; idx < NTOT;
       idx += gridDim.x * blockDim.x) {
    int i = idx;
    if (i < R0) {                                   // char embeds, gathered
      int t = i / CEMB, k = i % CEMB;
      CEbf[i] = (_Float16)char_emb[char_ids[t] * CEMB + k];
    } else if ((i -= R0) < R1) {                    // Wih_c
      WihCbf[i] = (_Float16)Wih_c[i];
    } else if ((i -= R1) < R2) {                    // X word part (padded)
      int s = i / XPAD, k = i % XPAD;
      float v = (k < WEMB) ? word_emb[word_ids[s] * WEMB + k] : 0.0f;
      Xbf[s * XK + k] = (_Float16)v;
    } else {                                        // Wih padded to 448 cols
      i -= R2;
      int g = i / XK, k = i % XK;
      float v;
      if (k < WEMB)      v = Wih[g * (WEMB + CHID) + k];
      else if (k < XPAD) v = 0.0f;
      else               v = Wih[g * (WEMB + CHID) + (k - 20)];  // k-320+300
      WihWbf[i] = (_Float16)v;
    }
  }
}

// ---------------------------------------------------------------------------
// WMMA fragment loaders (ISA 7.12.2 layouts, 16-bit elements, wave32)
//   A lane L: m = (L&15); VGPR 0-3 hold K = (L>>4)*8 + 0..7,
//             VGPR 4-7 hold K = 16 + (L>>4)*8 + 0..7          -> two b128 loads
//   B lane L: n = (L&15); same K layout as A                  -> two b128 loads
// ---------------------------------------------------------------------------
__device__ __forceinline__ v16h load_a_frag(const _Float16* __restrict__ row,
                                            int k0, int hsel) {
  v8h lo = *(const v8h*)(row + k0 + hsel * 8);
  v8h hi = *(const v8h*)(row + k0 + 16 + hsel * 8);
  v16h r;
#pragma unroll
  for (int i = 0; i < 8; ++i) { r[i] = lo[i]; r[8 + i] = hi[i]; }
  return r;
}
__device__ __forceinline__ v16h load_b_frag(const _Float16* __restrict__ row,
                                            int k0, int hsel) {
  v8h lo = *(const v8h*)(row + k0 + hsel * 8);
  v8h hi = *(const v8h*)(row + k0 + 16 + hsel * 8);
  v16h r;
#pragma unroll
  for (int i = 0; i < 8; ++i) { r[i] = lo[i]; r[8 + i] = hi[i]; }
  return r;
}

// ---------------------------------------------------------------------------
// Kernel 2: WMMA GEMM  D[M][N] = A[M][K] * Bt[N][K]^T  (f16 in, f32 acc)
// Register-blocked 16x32 per wave (one A frag feeds two B tiles), software-
// pipelined so loads for K-slice n+1 are in flight during the WMMAs of slice n
// (avoids the s_wait_loadcnt 0 stall seen in the 1-tile version).
// Requires: M % 16 == 0, N % 32 == 0, K % 32 == 0.
// ---------------------------------------------------------------------------
__global__ void wmma_gemm_kernel(const _Float16* __restrict__ A, int lda,
                                 const _Float16* __restrict__ Bt, int ldb,
                                 float* __restrict__ D, int ldd,
                                 int M, int N, int K) {
  const int lane  = threadIdx.x & 31;
  const int hsel  = lane >> 4;
  const int lr    = lane & 15;
  const int wave  = (blockIdx.x * blockDim.x + threadIdx.x) >> 5;
  const int nwav  = (gridDim.x * blockDim.x) >> 5;
  const int tM = M >> 4, tN2 = N >> 5, ntiles = tM * tN2;

  for (int tile = wave; tile < ntiles; tile += nwav) {
    const int tm = tile / tN2, tn = tile % tN2;
    const _Float16* arow  = A  + (size_t)(tm * 16 + lr) * lda;
    const _Float16* brow0 = Bt + (size_t)(tn * 32 + lr) * ldb;
    const _Float16* brow1 = Bt + (size_t)(tn * 32 + 16 + lr) * ldb;

    v8f acc0 = {}, acc1 = {};
    v16h a_c  = load_a_frag(arow, 0, hsel);
    v16h b0_c = load_b_frag(brow0, 0, hsel);
    v16h b1_c = load_b_frag(brow1, 0, hsel);

    int k0 = 0;
    for (; k0 + 32 < K; k0 += 32) {
      // issue next slice's loads before consuming the current one
      v16h a_n  = load_a_frag(arow, k0 + 32, hsel);
      v16h b0_n = load_b_frag(brow0, k0 + 32, hsel);
      v16h b1_n = load_b_frag(brow1, k0 + 32, hsel);
      if (k0 + 64 < K)
        __builtin_prefetch((const void*)(arow + k0 + 64), 0, 3);
      acc0 = __builtin_amdgcn_wmma_f32_16x16x32_f16(
          false, a_c, false, b0_c, (short)0, acc0, false, false);
      acc1 = __builtin_amdgcn_wmma_f32_16x16x32_f16(
          false, a_c, false, b1_c, (short)0, acc1, false, false);
      a_c = a_n; b0_c = b0_n; b1_c = b1_n;
    }
    acc0 = __builtin_amdgcn_wmma_f32_16x16x32_f16(
        false, a_c, false, b0_c, (short)0, acc0, false, false);
    acc1 = __builtin_amdgcn_wmma_f32_16x16x32_f16(
        false, a_c, false, b1_c, (short)0, acc1, false, false);

    float* drow = D + (size_t)(tm * 16 + hsel * 8) * ldd + tn * 32 + lr;
#pragma unroll
    for (int v = 0; v < 8; ++v) {
      drow[(size_t)v * ldd]      = acc0[v];
      drow[(size_t)v * ldd + 16] = acc1[v];
    }
  }
}

// ---------------------------------------------------------------------------
// Kernel 3: char LSTM recurrence (1 block, 512 threads, state in LDS)
//   g[j] = Gc[t][j] + bih_c[j] + bhh_c[j] + Whh_c[j,:]·hc   (float4 loads)
// After each word's 16 chars, emit hc into X columns [320,448) as f16.
// ---------------------------------------------------------------------------
__global__ void __launch_bounds__(GC4)
char_rec_kernel(const float* __restrict__ Gc,
                const float* __restrict__ Whh_c,
                const float* __restrict__ bih_c,
                const float* __restrict__ bhh_c,
                _Float16* __restrict__ Xbf) {
  __shared__ float hc[CHID], cc[CHID], gbuf[GC4];
  const int j = threadIdx.x;           // 0..511
  if (j < CHID) { hc[j] = 0.0f; cc[j] = 0.0f; }
  __syncthreads();
  const float bsum = bih_c[j] + bhh_c[j];
  const float4* wrow = (const float4*)(Whh_c + (size_t)j * CHID);
  const float4* hc4  = (const float4*)hc;
  for (int t = 0; t < TCH; ++t) {
    float acc = Gc[(size_t)t * GC4 + j] + bsum;
#pragma unroll 8
    for (int k = 0; k < CHID / 4; ++k) {
      float4 w = wrow[k];
      float4 h = hc4[k];
      acc = fmaf(w.x, h.x, acc);
      acc = fmaf(w.y, h.y, acc);
      acc = fmaf(w.z, h.z, acc);
      acc = fmaf(w.w, h.w, acc);
    }
    gbuf[j] = acc;
    __syncthreads();
    if (j < CHID) {
      float ig = sigm(gbuf[j]);
      float fg = sigm(gbuf[CHID + j]);
      float gg = tanhf(gbuf[2 * CHID + j]);
      float og = sigm(gbuf[3 * CHID + j]);
      float cn = fg * cc[j] + ig * gg;
      cc[j] = cn;
      float hn = og * tanhf(cn);
      hc[j] = hn;
      if ((t & (CPW - 1)) == CPW - 1)
        Xbf[(size_t)(t / CPW) * XK + XPAD + j] = (_Float16)hn;
    }
    __syncthreads();
  }
}

// ---------------------------------------------------------------------------
// Kernel 4: word LSTMCell recurrence (1 block, 1024 threads, 2 gate rows each)
// ---------------------------------------------------------------------------
__global__ void __launch_bounds__(1024)
word_rec_kernel(const float* __restrict__ Gw,
                const float* __restrict__ Whh,
                const float* __restrict__ bih,
                const float* __restrict__ bhh,
                float* __restrict__ hfin) {
  __shared__ float h[HID], cst[HID], gbuf[GW4];
  const int tid = threadIdx.x;         // 0..1023
  if (tid < HID) { h[tid] = 0.0f; cst[tid] = 0.0f; }
  __syncthreads();
  const float4* h4 = (const float4*)h;
  for (int s = 0; s < SEQ; ++s) {
#pragma unroll
    for (int r = 0; r < 2; ++r) {
      const int j = r * 1024 + tid;
      const float4* wrow = (const float4*)(Whh + (size_t)j * HID);
      float acc = Gw[(size_t)s * GW4 + j] + bih[j] + bhh[j];
#pragma unroll 8
      for (int k = 0; k < HID / 4; ++k) {
        float4 w = wrow[k];
        float4 hv = h4[k];
        acc = fmaf(w.x, hv.x, acc);
        acc = fmaf(w.y, hv.y, acc);
        acc = fmaf(w.z, hv.z, acc);
        acc = fmaf(w.w, hv.w, acc);
      }
      gbuf[j] = acc;
    }
    __syncthreads();
    if (tid < HID) {
      float ig = sigm(gbuf[tid]);
      float fg = sigm(gbuf[HID + tid]);
      float gg = tanhf(gbuf[2 * HID + tid]);
      float og = sigm(gbuf[3 * HID + tid]);
      float cn = fg * cst[tid] + ig * gg;
      cst[tid] = cn;
      h[tid] = og * tanhf(cn);
    }
    __syncthreads();
  }
  if (tid < HID) hfin[tid] = h[tid];
}

// ---------------------------------------------------------------------------
// Kernel 5: hidden = h + mean(hash_emb[hash_ids]);  out = Wout @ hidden + bout
// ---------------------------------------------------------------------------
__global__ void __launch_bounds__(HID)
finalize_kernel(const float* __restrict__ hfin,
                const int* __restrict__ hash_ids,
                const float* __restrict__ hash_emb,
                const float* __restrict__ Wout,
                const float* __restrict__ bout,
                float* __restrict__ out) {
  __shared__ float hidden[HID];
  const int tid = threadIdx.x;
  float m = 0.0f;
#pragma unroll
  for (int i = 0; i < NHASH; ++i) m += hash_emb[(size_t)hash_ids[i] * HID + tid];
  hidden[tid] = hfin[tid] + m * (1.0f / NHASH);
  __syncthreads();
  if (tid < TGT) {
    const float4* wrow = (const float4*)(Wout + (size_t)tid * HID);
    const float4* h4 = (const float4*)hidden;
    float acc = bout[tid];
#pragma unroll 8
    for (int k = 0; k < HID / 4; ++k) {
      float4 w = wrow[k];
      float4 hv = h4[k];
      acc = fmaf(w.x, hv.x, acc);
      acc = fmaf(w.y, hv.y, acc);
      acc = fmaf(w.z, hv.z, acc);
      acc = fmaf(w.w, hv.w, acc);
    }
    out[tid] = acc;
  }
}

// ---------------------------------------------------------------------------
// Launch
// ---------------------------------------------------------------------------
extern "C" void kernel_launch(void* const* d_in, const int* in_sizes, int n_in,
                              void* d_out, int out_size, void* d_ws, size_t ws_size,
                              hipStream_t stream) {
  (void)in_sizes; (void)n_in; (void)out_size; (void)ws_size;
  const int*   word_ids = (const int*)d_in[0];
  const int*   char_ids = (const int*)d_in[1];
  const int*   hash_ids = (const int*)d_in[2];
  const float* word_emb = (const float*)d_in[3];
  const float* char_emb = (const float*)d_in[4];
  const float* hash_emb = (const float*)d_in[5];
  const float* Wih_c    = (const float*)d_in[6];
  const float* Whh_c    = (const float*)d_in[7];
  const float* bih_c    = (const float*)d_in[8];
  const float* bhh_c    = (const float*)d_in[9];
  const float* Wih      = (const float*)d_in[10];
  const float* Whh      = (const float*)d_in[11];
  const float* bih      = (const float*)d_in[12];
  const float* bhh      = (const float*)d_in[13];
  const float* Wout     = (const float*)d_in[14];
  const float* bout     = (const float*)d_in[15];
  float* out = (float*)d_out;

  char* ws = (char*)d_ws;
  _Float16* CEbf   = (_Float16*)(ws + OFF_CEBF);
  _Float16* WihCbf = (_Float16*)(ws + OFF_WIHC);
  _Float16* Xbf    = (_Float16*)(ws + OFF_XBF);
  _Float16* WihWbf = (_Float16*)(ws + OFF_WIHW);
  float*    Gc     = (float*)(ws + OFF_GC);
  float*    Gw     = (float*)(ws + OFF_GW);
  float*    hfin   = (float*)(ws + OFF_HFIN);

  // 1. gather + convert (parallel)
  prep_kernel<<<512, 256, 0, stream>>>(word_ids, char_ids, word_emb, char_emb,
                                       Wih_c, Wih, CEbf, WihCbf, Xbf, WihWbf);

  // 2. Gc[1024][512] = CE[1024][64] x Wih_c^T   (WMMA, 1024 16x32 tiles)
  wmma_gemm_kernel<<<128, 256, 0, stream>>>(CEbf, CEMB, WihCbf, CEMB,
                                            Gc, GC4, TCH, GC4, CEMB);

  // 3. char LSTM recurrence (fills X[:,320:448] with hc per word)
  char_rec_kernel<<<1, GC4, 0, stream>>>(Gc, Whh_c, bih_c, bhh_c, Xbf);

  // 4. Gw[64][2048] = X[64][448] x Wih^T   (WMMA, 256 16x32 tiles)
  wmma_gemm_kernel<<<32, 256, 0, stream>>>(Xbf, XK, WihWbf, XK,
                                           Gw, GW4, SEQ, GW4, XK);

  // 5. word LSTMCell recurrence
  word_rec_kernel<<<1, 1024, 0, stream>>>(Gw, Whh, bih, bhh, hfin);

  // 6. hash mean + output head
  finalize_kernel<<<1, HID, 0, stream>>>(hfin, hash_ids, hash_emb, Wout, bout, out);
}
